// EdgeMLP_76390288327364
// MI455X (gfx1250) — matrix operand order repaired and checked by reading, not compiled, not run on hardware
//
#include <hip/hip_runtime.h>
#include <hip/hip_bf16.h>

typedef __attribute__((ext_vector_type(16))) __bf16 v16bf;
typedef __attribute__((ext_vector_type(8)))  __bf16 v8bf;
typedef __attribute__((ext_vector_type(8)))  float  v8f;

#define E_TOT   400000
#define N_NODES 100000
#define D_E     128
#define D_N     128
#define CAT_K   384          // D_E + 2*D_N
#define H_DIM   256
#define D_OUT   128

#define TILE_M  64           // edges per workgroup
#define CAT_P   392          // bf16 pitch for cat tile (784B, 16B aligned, bank-skewed)
#define H_P     264          // bf16 pitch for h tile  (528B)
#define Y_P     132          // f32 pitch for y tile   (528B, float4-aligned rows)

// ---------------------------------------------------------------------------
// Prep: transpose + downconvert weights to bf16 so WMMA B-fragments are
// contiguous per lane.  W1t: [H=256][K=384], W2t: [D_OUT=128][K=256].
// ---------------------------------------------------------------------------
__global__ void edgemlp_prep_kernel(const float* __restrict__ W1,
                                    const float* __restrict__ W2,
                                    __bf16* __restrict__ W1t,
                                    __bf16* __restrict__ W2t) {
    int i = blockIdx.x * 256 + threadIdx.x;
    if (i < H_DIM * CAT_K) {                // W1 is [384][256] row-major
        int n = i / CAT_K, k = i % CAT_K;
        W1t[i] = (__bf16)W1[k * H_DIM + n];
    }
    if (i < D_OUT * H_DIM) {                // W2 is [256][128] row-major
        int n = i / H_DIM, k = i % H_DIM;
        W2t[i] = (__bf16)W2[k * D_OUT + n];
    }
}

// A-fragment: 16x32 bf16, row-major source.  Contiguous k order, same as load_b_frag:
// lanes 0-15: M=lane, K = k0..k0+15
// lanes 16-31: M=lane-16, K = k0+16..k0+31
__device__ inline v16bf load_a_frag(const __bf16* base, int pitch,
                                    int row0, int k0, int lane) {
    int r  = row0 + (lane & 15);
    int kb = k0 + ((lane < 16) ? 0 : 16);
    const __bf16* p = base + r * pitch + kb;
    v8bf lo = *(const v8bf*)p;
    v8bf hi = *(const v8bf*)(p + 8);
    v16bf a;
#pragma unroll
    for (int i = 0; i < 8; ++i) { a[i] = lo[i]; a[i + 8] = hi[i]; }
    return a;
}

// B-fragment: 32x16 bf16 from transposed weights Wt[n][k].  Contiguous k order, same as A:
// lanes 0-15 hold K=k0..k0+15 of column n0+lane; lanes 16-31 hold K=k0+16..k0+31.
__device__ inline v16bf load_b_frag(const __bf16* __restrict__ Wt, int pitch,
                                    int n0, int k0, int lane) {
    int n  = n0 + (lane & 15);
    int kb = k0 + ((lane < 16) ? 0 : 16);
    return *(const v16bf*)(Wt + (size_t)n * pitch + kb);
}

// ---------------------------------------------------------------------------
// Fused: gather+concat -> GEMM1(bf16 WMMA) -> SiLU -> GEMM2 -> LayerNorm
// ---------------------------------------------------------------------------
__global__ void __launch_bounds__(256, 1)
edgemlp_fused_kernel(const float* __restrict__ efeat,
                     const float* __restrict__ nfeat,
                     const int*   __restrict__ srcI,
                     const int*   __restrict__ dstI,
                     const __bf16* __restrict__ W1t,
                     const float* __restrict__ b1,
                     const __bf16* __restrict__ W2t,
                     const float* __restrict__ b2,
                     const float* __restrict__ gamma,
                     const float* __restrict__ beta,
                     float* __restrict__ out) {
    __shared__ __align__(16) __bf16 catLds[TILE_M * CAT_P];
    __shared__ __align__(16) __bf16 hLds[TILE_M * H_P];
    __shared__ __align__(16) float  yLds[TILE_M * Y_P];
    __shared__ int   sIdx[TILE_M];
    __shared__ int   dIdx[TILE_M];
    __shared__ float b1Lds[H_DIM];
    __shared__ float b2Lds[D_OUT];
    __shared__ float gLds[D_OUT];
    __shared__ float betaLds[D_OUT];

    const int tid   = threadIdx.x;
    const int wave  = tid >> 5;
    const int lane  = tid & 31;
    const int ln    = lane & 15;
    const int eBase = blockIdx.x * TILE_M;

    // ---- preload indices / bias / LN params -------------------------------
    if (tid < TILE_M) {
        sIdx[tid] = srcI[eBase + tid];
        dIdx[tid] = dstI[eBase + tid];
    }
    if (tid < H_DIM) b1Lds[tid] = b1[tid];
    if (tid < D_OUT) {
        b2Lds[tid]   = b2[tid];
        gLds[tid]    = gamma[tid];
        betaLds[tid] = beta[tid];
    }
    __syncthreads();

    // ---- stage 1: gather + concat + f32->bf16 into LDS --------------------
    // 64 rows x 96 float4-chunks; chunk c4: [0,32)=efeat, [32,64)=nfeat[src], [64,96)=nfeat[dst]
#pragma unroll 4
    for (int i = tid; i < TILE_M * 96; i += 256) {
        int row = i / 96;
        int c4  = i - row * 96;
        const float* sp;
        if (c4 < 32)      sp = efeat + (size_t)(eBase + row) * D_E + c4 * 4;
        else if (c4 < 64) sp = nfeat + (size_t)sIdx[row] * D_N + (c4 - 32) * 4;
        else              sp = nfeat + (size_t)dIdx[row] * D_N + (c4 - 64) * 4;
        float4 v = *(const float4*)sp;
        __bf16* d = &catLds[row * CAT_P + c4 * 4];
        d[0] = (__bf16)v.x; d[1] = (__bf16)v.y;
        d[2] = (__bf16)v.z; d[3] = (__bf16)v.w;
    }
    __syncthreads();

    // ---- GEMM1: [64 x 384] x [384 x 256] -> h, wave owns 64x32 of N -------
    const int nBase1 = wave * 32;
    v8f c1[2][4];
#pragma unroll
    for (int nt = 0; nt < 2; ++nt)
#pragma unroll
        for (int mt = 0; mt < 4; ++mt) c1[nt][mt] = (v8f)0.0f;

    for (int k0 = 0; k0 < CAT_K; k0 += 32) {
        v16bf a[4];
#pragma unroll
        for (int mt = 0; mt < 4; ++mt)
            a[mt] = load_a_frag(catLds, CAT_P, mt * 16, k0, lane);
#pragma unroll
        for (int nt = 0; nt < 2; ++nt) {
            v16bf b = load_b_frag(W1t, CAT_K, nBase1 + nt * 16, k0, lane);
#pragma unroll
            for (int mt = 0; mt < 4; ++mt)
                c1[nt][mt] = __builtin_amdgcn_wmma_f32_16x16x32_bf16(
                    false, a[mt], false, b, (short)0, c1[nt][mt], false, false);
        }
    }

    // ---- bias + SiLU, bf16 into hLds --------------------------------------
    // sigmoid via v_exp_f32 + v_rcp_f32 (avoid the IEEE division expansion)
#pragma unroll
    for (int nt = 0; nt < 2; ++nt) {
        int col = nBase1 + nt * 16 + ln;
        float bb = b1Lds[col];
#pragma unroll
        for (int mt = 0; mt < 4; ++mt) {
            int rbase = mt * 16 + ((lane < 16) ? 0 : 8);
#pragma unroll
            for (int r = 0; r < 8; ++r) {
                float x = c1[nt][mt][r] + bb;
                float s = x * __builtin_amdgcn_rcpf(1.0f + __expf(-x)); // SiLU
                hLds[(rbase + r) * H_P + col] = (__bf16)s;
            }
        }
    }
    __syncthreads();

    // ---- GEMM2: [64 x 256] x [256 x 128] -> y, wave owns 64x16 of N -------
    const int nBase2 = wave * 16;
    v8f c2[4];
#pragma unroll
    for (int mt = 0; mt < 4; ++mt) c2[mt] = (v8f)0.0f;

    for (int k0 = 0; k0 < H_DIM; k0 += 32) {
        v16bf b = load_b_frag(W2t, H_DIM, nBase2, k0, lane);
#pragma unroll
        for (int mt = 0; mt < 4; ++mt) {
            v16bf a = load_a_frag(hLds, H_P, mt * 16, k0, lane);
            c2[mt] = __builtin_amdgcn_wmma_f32_16x16x32_bf16(
                false, a, false, b, (short)0, c2[mt], false, false);
        }
    }

    // ---- y + bias -> f32 LDS ----------------------------------------------
    {
        int col = nBase2 + ln;
        float bb = b2Lds[col];
#pragma unroll
        for (int mt = 0; mt < 4; ++mt) {
            int rbase = mt * 16 + ((lane < 16) ? 0 : 8);
#pragma unroll
            for (int r = 0; r < 8; ++r)
                yLds[(rbase + r) * Y_P + col] = c2[mt][r] + bb;
        }
    }
    __syncthreads();

    // ---- LayerNorm over D_OUT=128, one thread per row ---------------------
    if (tid < TILE_M) {
        float* yr = &yLds[tid * Y_P];
        float s = 0.0f, s2 = 0.0f;
#pragma unroll
        for (int i4 = 0; i4 < D_OUT / 4; ++i4) {
            float4 v = *(const float4*)&yr[i4 * 4];
            s  += v.x + v.y + v.z + v.w;
            s2 += v.x * v.x + v.y * v.y + v.z * v.z + v.w * v.w;
        }
        float mu  = s * (1.0f / D_OUT);
        float var = s2 * (1.0f / D_OUT) - mu * mu;
        float inv = __builtin_amdgcn_rsqf(var + 1e-5f);
#pragma unroll 8
        for (int i = 0; i < D_OUT; ++i)
            yr[i] = (yr[i] - mu) * inv * gLds[i] + betaLds[i];
    }
    __syncthreads();

    // ---- coalesced float4 copy-out ----------------------------------------
#pragma unroll
    for (int i = tid; i < TILE_M * 32; i += 256) {
        int row = i >> 5;
        int c4  = i & 31;
        float4 v = *(const float4*)&yLds[row * Y_P + c4 * 4];
        *(float4*)(out + (size_t)(eBase + row) * D_OUT + c4 * 4) = v;
    }
}

// ---------------------------------------------------------------------------
extern "C" void kernel_launch(void* const* d_in, const int* in_sizes, int n_in,
                              void* d_out, int out_size, void* d_ws, size_t ws_size,
                              hipStream_t stream) {
    const float* efeat = (const float*)d_in[0];
    const float* nfeat = (const float*)d_in[1];
    const int*   srcI  = (const int*)d_in[2];
    const int*   dstI  = (const int*)d_in[3];
    const float* W1    = (const float*)d_in[4];
    const float* b1    = (const float*)d_in[5];
    const float* W2    = (const float*)d_in[6];
    const float* b2    = (const float*)d_in[7];
    const float* gam   = (const float*)d_in[8];
    const float* bet   = (const float*)d_in[9];
    float* out = (float*)d_out;

    __bf16* W1t = (__bf16*)d_ws;                                 // 256*384*2 = 196608 B
    __bf16* W2t = (__bf16*)((char*)d_ws + H_DIM * CAT_K * 2);    // 128*256*2 =  65536 B

    edgemlp_prep_kernel<<<(H_DIM * CAT_K + 255) / 256, 256, 0, stream>>>(W1, W2, W1t, W2t);

    edgemlp_fused_kernel<<<E_TOT / TILE_M, 256, 0, stream>>>(
        efeat, nfeat, srcI, dstI, W1t, b1, W2t, b2, gam, bet, out);
}
